// BiMamba_77446850282154
// MI455X (gfx1250) — matrix operand order repaired and checked by reading, not compiled, not run on hardware
//
#include <hip/hip_runtime.h>
#include <hip/hip_bf16.h>

#define D_IN    256
#define D_INT   512
#define N_ST    16
#define DT_RANK 16
#define D_CONV  4
#define B_SZ    4
#define SEQ_L   1024
#define NROWS   (B_SZ * SEQ_L)            /* 4096 */
#define XDBL_C  (DT_RANK + 4 * N_ST)      /* 80   */
#define CHUNK   16

/* LDS-staged GEMM tiling */
#define BM  128   /* 8 waves x 16 rows  */
#define BN  64    /* 4 N-tiles per wave */
#define BK  32
#define BKP 56    /* padded LDS row: 112B stride -> 16B aligned, conflict-free */

typedef __attribute__((ext_vector_type(16))) _Float16 v16h;
typedef __attribute__((ext_vector_type(8)))  float    v8f;

__device__ __forceinline__ float silu_f(float x) { return x / (1.0f + __expf(-x)); }

// ---------------------------------------------------------------------------
// LDS-staged WMMA GEMM: C[MxN] = A[MxK] * B[KxN], row-major f32, f16 compute.
// Block = 8 waves, block tile 128x64, K-step 32.
//  - B panel (32x64) staged to LDS transposed as f16 [n][k] (padded rows);
//    shared by all 8 waves; fragment read = 2x ds_load_b128 per tile.
//  - A fragment per wave: two contiguous 8-float groups per lane
//    (global_load_b128 + v_cvt_pk_f16_f32), reused across 4 WMMAs.
// Requires M%128==0, N%64==0, K%32==0.
// ---------------------------------------------------------------------------
__global__ void __launch_bounds__(256)
wmma_gemm_lds(const float* __restrict__ A, const float* __restrict__ B,
              float* __restrict__ C, int M, int N, int K)
{
  const int lane = threadIdx.x & 31;
  const int wave = threadIdx.x >> 5;   // M sub-tile 0..7
  const int half = lane >> 4;          // 0 or 1
  const int l16  = lane & 15;

  const int bm = blockIdx.x;           // M / 128
  const int bn = blockIdx.y;           // N / 64

  __shared__ _Float16 Bs[BN][BKP];     // transposed B panel, f16

  const float* Arow = A + (size_t)(bm * BM + wave * 16 + l16) * K;
  const int ka0 = half * 16;           // K group 0 base (0 or 16)
  const int ka1 = half * 16 + 8;       // K group 1 base (8 or 24)

  // B staging role: thread t -> column n = t&63, k-range (t>>6)*8..+7
  const int sn = threadIdx.x & 63;
  const int sk = (threadIdx.x >> 6) * 8;
  const float* Bstage = B + (size_t)(bn * BN + sn);

  v8f acc[4];
#pragma unroll
  for (int j = 0; j < 4; ++j) acc[j] = (v8f){};

  for (int kk = 0; kk < K; kk += BK) {
    // ---- stage B panel (coalesced along n; strided along k) ----
    {
      const float* Bp = Bstage + (size_t)(kk + sk) * N;
#pragma unroll
      for (int i = 0; i < 8; ++i)
        Bs[sn][sk + i] = (_Float16)Bp[(size_t)i * N];
    }
    if (kk + BK < K) {
      __builtin_prefetch(Arow + kk + BK, 0, 3);                       // global_prefetch_b8
      __builtin_prefetch(Bstage + (size_t)(kk + BK + sk) * N, 0, 3);
    }
    __syncthreads();

    // ---- A fragment (16x32 f16), contiguous k = 16*half + i per lane (same order as B) ----
    v16h a;
#pragma unroll
    for (int i = 0; i < 8; ++i) {
      a[i]     = (_Float16)Arow[kk + ka0 + i];
      a[8 + i] = (_Float16)Arow[kk + ka1 + i];
    }

    // ---- 4 N-tiles: B fragment from LDS (2x ds_load_b128), WMMA ----
#pragma unroll
    for (int j = 0; j < 4; ++j) {
      const _Float16* bp = &Bs[j * 16 + l16][half * 16];
      v16h b;
#pragma unroll
      for (int i = 0; i < 16; ++i) b[i] = bp[i];
      acc[j] = __builtin_amdgcn_wmma_f32_16x16x32_f16(
          /*neg_a=*/false, a, /*neg_b=*/false, b,
          /*c_mod=*/(short)0, acc[j], /*reuse_a=*/false, /*reuse_b=*/false);
    }
    __syncthreads();
  }

  // ---- store 16x64 per wave ----
  float* Crow = C + (size_t)(bm * BM + wave * 16 + half * 8) * N + bn * BN + l16;
#pragma unroll
  for (int j = 0; j < 4; ++j) {
#pragma unroll
    for (int i = 0; i < 8; ++i)
      Crow[(size_t)i * N + j * 16] = acc[j][i];
  }
}

// ---------------------------------------------------------------------------
// Simple per-tile WMMA GEMM (for N=80 x_dbl GEMM; any M%16, N%16, K%32).
// ---------------------------------------------------------------------------
__global__ void __launch_bounds__(256)
wmma_gemm_f32f16(const float* __restrict__ A, const float* __restrict__ B,
                 float* __restrict__ C, int M, int N, int K)
{
  const int lane = threadIdx.x & 31;
  const int wave = threadIdx.x >> 5;
  const int half = lane >> 4;
  const int l16  = lane & 15;
  const int ntN  = N >> 4;
  const int ntM  = M >> 4;
  const int tile = blockIdx.x * 8 + wave;   // uniform within wave
  if (tile >= ntM * ntN) return;
  const int tm = tile / ntN;
  const int tn = tile - tm * ntN;

  const float* Arow  = A + (size_t)(tm * 16 + l16) * K;
  const float* Bbase = B + (size_t)(tn * 16 + l16);
  const int ka0 = half * 16;
  const int ka1 = half * 16 + 8;

  v8f acc = {};
  for (int kk = 0; kk < K; kk += 32) {
    v16h a, b;
#pragma unroll
    for (int i = 0; i < 8; ++i) {
      a[i]     = (_Float16)Arow[kk + ka0 + i];
      a[8 + i] = (_Float16)Arow[kk + ka1 + i];
    }
    const float* Bp = Bbase + (size_t)(kk + half * 16) * N;
#pragma unroll
    for (int i = 0; i < 16; ++i)
      b[i] = (_Float16)Bp[(size_t)i * N];
    acc = __builtin_amdgcn_wmma_f32_16x16x32_f16(
        false, a, false, b, (short)0, acc, false, false);
  }
  float* Cp = C + (size_t)(tm * 16 + half * 8) * N + tn * 16 + l16;
#pragma unroll
  for (int i = 0; i < 8; ++i)
    Cp[(size_t)i * N] = acc[i];
}

// ---------------------------------------------------------------------------
// Depthwise conv (k=4, pad (1,2)) over L on x = x_and_res[:, :512], + SiLU.
// ---------------------------------------------------------------------------
__global__ void __launch_bounds__(256)
conv_silu_kernel(const float* __restrict__ xar, const float* __restrict__ ck,
                 const float* __restrict__ cb, float* __restrict__ xs)
{
  const int idx = blockIdx.x * 256 + threadIdx.x;
  if (idx >= NROWS * D_INT) return;
  const int c   = idx & (D_INT - 1);
  const int row = idx >> 9;
  const int l   = row & (SEQ_L - 1);
  const int b   = row >> 10;
  float acc = cb[c];
#pragma unroll
  for (int k = 0; k < D_CONV; ++k) {
    const int ll = l - 1 + k;
    if (ll >= 0 && ll < SEQ_L)
      acc += xar[(size_t)(b * SEQ_L + ll) * (2 * D_INT) + c] * ck[k * D_INT + c];
  }
  xs[idx] = silu_f(acc);
}

// ---------------------------------------------------------------------------
// delta = softplus(x_dbl[:, :16] @ W_dt + b_dt).  K=16 -> plain VALU dot.
// ---------------------------------------------------------------------------
__global__ void __launch_bounds__(256)
delta_kernel(const float* __restrict__ xdbl, const float* __restrict__ Wdt,
             const float* __restrict__ bdt, float* __restrict__ dlt)
{
  const int idx = blockIdx.x * 256 + threadIdx.x;
  if (idx >= NROWS * D_INT) return;
  const int d   = idx & (D_INT - 1);
  const int row = idx >> 9;
  float acc = bdt[d];
  const float* xr = xdbl + (size_t)row * XDBL_C;
#pragma unroll
  for (int r = 0; r < DT_RANK; ++r)
    acc += xr[r] * Wdt[r * D_INT + d];
  dlt[idx] = (acc > 20.0f) ? acc : __logf(1.0f + __expf(acc));
}

// ---------------------------------------------------------------------------
// Bidirectional selective scan.  One lane per channel d; h[16] in registers.
// B/C rows (shared across all d) staged through LDS in CHUNK-step tiles.
// bwd: u/delta/output use t = L-1-s, B/C use s (reference does NOT flip B/C).
// Fuses + u*D and * silu(res) epilogue into ymul[:, dir*512 + d].
// ---------------------------------------------------------------------------
__global__ void __launch_bounds__(256)
scan_kernel(const float* __restrict__ xs,   const float* __restrict__ dlt,
            const float* __restrict__ xdbl, const float* __restrict__ A_log,
            const float* __restrict__ D_param, const float* __restrict__ xar,
            float* __restrict__ ymul)
{
  const int d   = blockIdx.x * 256 + threadIdx.x;   // 0..511
  const int b   = blockIdx.y;                       // 0..3
  const int dir = blockIdx.z;                       // 0=fwd, 1=bwd

  float Ar[N_ST];
#pragma unroll
  for (int n = 0; n < N_ST; ++n) Ar[n] = -__expf(A_log[d * N_ST + n]);
  const float Dp = D_param[d];
  float h[N_ST];
#pragma unroll
  for (int n = 0; n < N_ST; ++n) h[n] = 0.0f;

  __shared__ float sB[CHUNK][N_ST];
  __shared__ float sC[CHUNK][N_ST];
  const int bcol = DT_RANK + dir * N_ST;             // Bm slice base column
  const int ccol = DT_RANK + 2 * N_ST + dir * N_ST;  // Cm slice base column

  for (int s0 = 0; s0 < SEQ_L; s0 += CHUNK) {
    for (int idx = threadIdx.x; idx < CHUNK * 2 * N_ST; idx += 256) {
      const int step = idx >> 5;
      const int j    = idx & 31;
      const size_t r = (size_t)(b * SEQ_L + s0 + step) * XDBL_C;
      if (j < N_ST) sB[step][j]        = xdbl[r + bcol + j];
      else          sC[step][j - N_ST] = xdbl[r + ccol + (j - N_ST)];
    }
    __syncthreads();
    for (int ss = 0; ss < CHUNK; ++ss) {
      const int s = s0 + ss;
      const int t = dir ? (SEQ_L - 1 - s) : s;       // time index (u/delta/out)
      const size_t rowt = (size_t)(b * SEQ_L + t);
      const float u  = xs[rowt * D_INT + d];
      const float dt = dlt[rowt * D_INT + d];
      const float du = dt * u;
      float y = 0.0f;
#pragma unroll
      for (int n = 0; n < N_ST; ++n) {
        h[n] = __expf(dt * Ar[n]) * h[n] + du * sB[ss][n];
        y += h[n] * sC[ss][n];
      }
      y += u * Dp;
      const float r = xar[rowt * (2 * D_INT) + D_INT + d];   // res channel d
      ymul[rowt * (2 * D_INT) + dir * D_INT + d] = y * silu_f(r);
    }
    __syncthreads();
  }
}

// ---------------------------------------------------------------------------
extern "C" void kernel_launch(void* const* d_in, const int* in_sizes, int n_in,
                              void* d_out, int out_size, void* d_ws, size_t ws_size,
                              hipStream_t stream) {
  const float* inputs  = (const float*)d_in[0];
  const float* W_in    = (const float*)d_in[1];
  const float* conv_k  = (const float*)d_in[2];
  const float* conv_b  = (const float*)d_in[3];
  const float* W_x     = (const float*)d_in[4];
  const float* W_dt    = (const float*)d_in[5];
  const float* b_dt    = (const float*)d_in[6];
  const float* A_log   = (const float*)d_in[7];
  const float* D_param = (const float*)d_in[8];
  const float* W_out   = (const float*)d_in[9];
  float* out = (float*)d_out;

  float* ws   = (float*)d_ws;
  float* xar  = ws;                                   // 4096 x 1024
  float* xs   = xar  + (size_t)NROWS * (2 * D_INT);   // 4096 x 512
  float* xdbl = xs   + (size_t)NROWS * D_INT;         // 4096 x 80
  float* dlt  = xdbl + (size_t)NROWS * XDBL_C;        // 4096 x 512
  float* ymul = dlt  + (size_t)NROWS * D_INT;         // 4096 x 1024

  // 1) x_and_res = inputs @ W_in            (4096 x 1024, K=256)
  {
    dim3 grid(NROWS / BM, (2 * D_INT) / BN);
    wmma_gemm_lds<<<grid, 256, 0, stream>>>(inputs, W_in, xar,
                                            NROWS, 2 * D_INT, D_IN);
  }
  // 2) depthwise conv + SiLU -> xs
  conv_silu_kernel<<<(NROWS * D_INT) / 256, 256, 0, stream>>>(xar, conv_k, conv_b, xs);
  // 3) x_dbl = xs @ W_x                     (4096 x 80, K=512) -- N%64 != 0
  {
    const int tiles = (NROWS / 16) * (XDBL_C / 16);
    wmma_gemm_f32f16<<<(tiles + 7) / 8, 256, 0, stream>>>(
        xs, W_x, xdbl, NROWS, XDBL_C, D_INT);
  }
  // 4) delta = softplus(x_dbl[:, :16] @ W_dt + b_dt)
  delta_kernel<<<(NROWS * D_INT) / 256, 256, 0, stream>>>(xdbl, W_dt, b_dt, dlt);
  // 5) bidirectional scan + u*D + silu(res) gate -> ymul (4096 x 1024)
  {
    dim3 grid(D_INT / 256, B_SZ, 2);
    scan_kernel<<<grid, 256, 0, stream>>>(xs, dlt, xdbl, A_log, D_param, xar, ymul);
  }
  // 6) out = ymul @ W_out                   (4096 x 256, K=1024)
  {
    dim3 grid(NROWS / BM, D_IN / BN);
    wmma_gemm_lds<<<grid, 256, 0, stream>>>(ymul, W_out, out,
                                            NROWS, D_IN, 2 * D_INT);
  }
}
